// SvarogAttention_49821620634181
// MI455X (gfx1250) — matrix operand order repaired and checked by reading, not compiled, not run on hardware
//
#include <hip/hip_runtime.h>
#include <hip/hip_bf16.h>
#include <cstdint>

// MI455X / gfx1250, wave32. All matmuls via v_wmma_f32_16x16x32_bf16.
// Global->LDS via GLOBAL_LOAD_ASYNC_TO_LDS_B128 (ASYNCcnt), double-buffered
// in the GEMM so DMA overlaps WMMA. GEMM wave tile = 32x128 (16 WMMA/k-step,
// each B fragment reused by 2 WMMAs).
// Workspace (bf16/ushort elements), ~117.4 MB total:
//   x_bf[4096*2048] wq_bf wk_bf wv_bf wo_bf[2048*2048 each]
//   q_bf k_bf [B,nh,S,hd]   vT_bf [B,nh,hd,S]   ao_bf [B,S,H]

typedef __attribute__((ext_vector_type(16))) __bf16       v16bf;
typedef __attribute__((ext_vector_type(8)))  float        v8f;
typedef __attribute__((ext_vector_type(4)))  unsigned int u32x4;

union FragBF { v16bf v; u32x4 q[2]; };

__device__ __forceinline__ unsigned short f2bf(float f) {
  union { float f; unsigned u; } c; c.f = f;
  unsigned r = c.u + 0x7FFFu + ((c.u >> 16) & 1u);   // round-to-nearest-even
  return (unsigned short)(r >> 16);
}

__device__ __forceinline__ v8f wmma_bf16(const FragBF& a, const FragBF& b, v8f c) {
  return __builtin_amdgcn_wmma_f32_16x16x32_bf16(false, a.v, false, b.v,
                                                 (short)0, c, false, false);
}

// A-matrix 16x32 bf16 fragment (ISA 7.12.2): lane row = lane&15;
// dwords 0-3 hold K = hi*8+0..7, dwords 4-7 hold K = 16+hi*8+0..7.
__device__ __forceinline__ FragBF loadA(const unsigned short* rowp, int hi) {
  FragBF f;
  f.q[0] = *(const u32x4*)(rowp + hi * 8);
  f.q[1] = *(const u32x4*)(rowp + 16 + hi * 8);
  return f;
}
// B-matrix 32x16 bf16 fragment: lane col = lane&15; dwords 0-3 hold
// K = hi*8+0..7, dwords 4-7 hold K = 16+hi*8+0..7 (same as A).
__device__ __forceinline__ FragBF loadB(const unsigned short* rowp, int hi) {
  FragBF f;
  f.q[0] = *(const u32x4*)(rowp + hi * 8);
  f.q[1] = *(const u32x4*)(rowp + 16 + hi * 8);
  return f;
}

// Async DMA: global (16B) -> LDS, tracked by ASYNCcnt. GV addressing mode.
// LDS byte address = low 32 bits of generic shared pointer (ISA aperture rule).
__device__ __forceinline__ void async_g2l_b128(void* lds, const void* gptr) {
  unsigned la = (unsigned)(uintptr_t)lds;
  unsigned long long ga = (unsigned long long)(uintptr_t)gptr;
  asm volatile("global_load_async_to_lds_b128 %0, %1, off"
               :: "v"(la), "v"(ga) : "memory");
}
__device__ __forceinline__ void wait_async0() {
  asm volatile("s_wait_asynccnt 0x0" ::: "memory");
}
__device__ __forceinline__ void wait_async6() {
  asm volatile("s_wait_asynccnt 0x6" ::: "memory");
}

// fp32 -> bf16, 8 elements per thread (two b128 loads, one b128 store).
__global__ void svarog_cvt_bf16(const float* __restrict__ in,
                                unsigned short* __restrict__ out, int n8) {
  int i = blockIdx.x * blockDim.x + threadIdx.x;
  if (i >= n8) return;
  const float4 a = ((const float4*)in)[i * 2];
  const float4 b = ((const float4*)in)[i * 2 + 1];
  union { unsigned short h[8]; u32x4 v; } p;
  p.h[0] = f2bf(a.x); p.h[1] = f2bf(a.y); p.h[2] = f2bf(a.z); p.h[3] = f2bf(a.w);
  p.h[4] = f2bf(b.x); p.h[5] = f2bf(b.y); p.h[6] = f2bf(b.z); p.h[7] = f2bf(b.w);
  ((u32x4*)out)[i] = p.v;
}

// C[m,n] = sum_k X[m,k] * W[n,k]  (i.e. X @ W^T), X:[M,2048] W:[2048,2048]
// Block tile 256x128, 8 waves, wave tile 32x128.
// mode 0: fp32 out [M,2048]
// mode 1: Q proj -> RoPE + 1/sqrt(hd) -> bf16 [B,nh,S,hd]
// mode 2: K proj -> RoPE             -> bf16 [B,nh,S,hd]
// mode 3: V proj -> bf16 transposed     [B,nh,hd,S]
__global__ __launch_bounds__(256)
void svarog_gemm_wmma(const unsigned short* __restrict__ X,
                      const unsigned short* __restrict__ W,
                      unsigned short* __restrict__ outBf,
                      float* __restrict__ outF, int mode) {
  const int H = 2048, S = 2048, NH = 16, HD = 128;
  // double-buffered tiles, stride 40 (16B-aligned, bank-spread): 60 KB
  __shared__ __align__(16) unsigned short Alds[2][256 * 40];
  __shared__ __align__(16) unsigned short Blds[2][128 * 40];
  const int tid = threadIdx.x;
  const int wave = tid >> 5, lane = tid & 31, n16 = lane & 15, hi = lane >> 4;
  const int tN = blockIdx.x, tM = blockIdx.y;

  auto issue_tile = [&](int buf, int k0) {
    for (int j = tid; j < 1024; j += 256) {    // A: 256 rows x 32 bf16
      int row = j >> 2, c8 = (j & 3) << 3;
      async_g2l_b128(&Alds[buf][row * 40 + c8],
                     &X[(size_t)(tM * 256 + row) * H + k0 + c8]);
    }
    for (int j = tid; j < 512; j += 256) {     // B: 128 rows x 32 bf16
      int row = j >> 2, c8 = (j & 3) << 3;
      async_g2l_b128(&Blds[buf][row * 40 + c8],
                     &W[(size_t)(tN * 128 + row) * H + k0 + c8]);
    }
  };

  const v8f vzero = {0.f, 0.f, 0.f, 0.f, 0.f, 0.f, 0.f, 0.f};
  v8f acc[2][8];
#pragma unroll
  for (int mt = 0; mt < 2; mt++)
#pragma unroll
    for (int ct = 0; ct < 8; ct++) acc[mt][ct] = vzero;

  issue_tile(0, 0);
  int cur = 0;
  for (int k0 = 0; k0 < H; k0 += 32) {
    if (k0 + 32 < H) {
      issue_tile(cur ^ 1, k0 + 32);   // DMA next tile while computing this one
      wait_async6();                  // 6 newest outstanding = next tile only
    } else {
      wait_async0();
    }
    __syncthreads();
    FragBF a0 = loadA(&Alds[cur][(wave * 32 + n16) * 40], hi);
    FragBF a1 = loadA(&Alds[cur][(wave * 32 + 16 + n16) * 40], hi);
#pragma unroll
    for (int ct = 0; ct < 8; ct++) {
      FragBF b = loadB(&Blds[cur][(ct * 16 + n16) * 40], hi);
      acc[0][ct] = wmma_bf16(a0, b, acc[0][ct]);
      acc[1][ct] = wmma_bf16(a1, b, acc[1][ct]);
    }
    __syncthreads();                  // all waves done reading before overwrite
    cur ^= 1;
  }

  if (mode == 0) {
#pragma unroll
    for (int mt = 0; mt < 2; mt++)
#pragma unroll
      for (int ct = 0; ct < 8; ct++)
#pragma unroll
        for (int r = 0; r < 8; r++) {
          int m = tM * 256 + wave * 32 + mt * 16 + r + 8 * hi;
          int n = tN * 128 + ct * 16 + n16;
          outF[(size_t)m * H + n] = acc[mt][ct][r];
        }
  } else if (mode == 3) {
#pragma unroll
    for (int mt = 0; mt < 2; mt++)
#pragma unroll
      for (int ct = 0; ct < 8; ct++)
#pragma unroll
        for (int r = 0; r < 8; r++) {
          int m = tM * 256 + wave * 32 + mt * 16 + r + 8 * hi;
          int bb = m >> 11, s = m & 2047;
          int d = ct * 16 + n16;
          outBf[(((size_t)bb * NH + tN) * HD + d) * S + s] = f2bf(acc[mt][ct][r]);
        }
  } else {
    const float qscale = (mode == 1) ? 0.08838834764831845f : 1.0f;
#pragma unroll
    for (int mt = 0; mt < 2; mt++)
#pragma unroll
      for (int ct = 0; ct < 8; ct++)
#pragma unroll
        for (int r = 0; r < 8; r++) {
          int m = tM * 256 + wave * 32 + mt * 16 + r + 8 * hi;
          int bb = m >> 11, s = m & 2047;
          int d = ct * 16 + n16;
          // RoPE: angle = s * 10000^{-(d%64)/64}
          float ang = (float)s * __expf((float)(d & 63) * (-9.210340372f / 64.0f));
          float cs = __cosf(ang), sn = __sinf(ang);
          int ip = (ct < 4) ? (ct + 4) : (ct - 4);  // rotate-half partner tile
          float pv = acc[mt][ip][r];
          float rot = (ct < 4) ? -pv : pv;
          float val = (acc[mt][ct][r] * cs + rot * sn) * qscale;
          outBf[(((size_t)bb * NH + tN) * S + s) * HD + d] = f2bf(val);
        }
  }
}

// Flash attention: block = (b,h) x 128 q-rows; 8 waves x 16 rows each.
// 64-wide KV chunks DMA'd to LDS (async); online softmax; P->LDS->A-frag.
__global__ __launch_bounds__(256)
void svarog_flash_wmma(const unsigned short* __restrict__ Q,
                       const unsigned short* __restrict__ K,
                       const unsigned short* __restrict__ Vt,
                       const float* __restrict__ mask,
                       unsigned short* __restrict__ O) {
  const int S = 2048, HD = 128, H = 2048;
  __shared__ __align__(16) unsigned short Klds[64 * 136];
  __shared__ __align__(16) unsigned short Vtlds[128 * 72];
  __shared__ __align__(16) unsigned short Plds[8 * 16 * 72];
  const int tid = threadIdx.x, wave = tid >> 5, lane = tid & 31;
  const int n16 = lane & 15, hi = lane >> 4;
  const int bh = blockIdx.y, b = bh >> 4, h = bh & 15;
  const int qr0 = blockIdx.x * 128 + wave * 16;

  const unsigned short* Qbase = Q + ((size_t)bh * S + qr0) * HD;
  const unsigned short* Kbase = K + (size_t)bh * S * HD;
  const unsigned short* Vbase = Vt + (size_t)bh * HD * S;

  FragBF aq[4];
#pragma unroll
  for (int ks = 0; ks < 4; ks++) {
    const unsigned short* p = Qbase + n16 * HD + ks * 32;
    aq[ks].q[0] = *(const u32x4*)(p + hi * 8);
    aq[ks].q[1] = *(const u32x4*)(p + 16 + hi * 8);
  }

  const v8f vzero = {0.f, 0.f, 0.f, 0.f, 0.f, 0.f, 0.f, 0.f};
  v8f o[8];
  float mrow[8], lrow[8];
#pragma unroll
  for (int dt = 0; dt < 8; dt++) o[dt] = vzero;
#pragma unroll
  for (int r = 0; r < 8; r++) { mrow[r] = -3.0e38f; lrow[r] = 0.f; }

  unsigned short* Pw = &Plds[wave * 16 * 72];

  for (int kv0 = 0; kv0 < S; kv0 += 64) {
    const unsigned short* kp = Kbase + (size_t)kv0 * HD;
    const unsigned short* vp = Vbase + kv0;
    for (int j = tid; j < 1024; j += 256) {
      { int row = j >> 4, c8 = (j & 15) << 3;               // K: 64x128
        async_g2l_b128(&Klds[row * 136 + c8], &kp[row * HD + c8]); }
      { int row = j >> 3, c8 = (j & 7) << 3;                // Vt: 128x64
        async_g2l_b128(&Vtlds[row * 72 + c8], &vp[(size_t)row * S + c8]); }
    }
    if (kv0 + 64 < S)                                        // global_prefetch_b8
      __builtin_prefetch(kp + 64 * HD + tid * 32, 0, 1);
    wait_async0();
    __syncthreads();

    // scores 16x64: per k-step load 4 B-frags, then 4 independent WMMAs
    v8f sc[4];
#pragma unroll
    for (int ct = 0; ct < 4; ct++) sc[ct] = vzero;
#pragma unroll
    for (int ks = 0; ks < 4; ks++) {
      FragBF bk[4];
#pragma unroll
      for (int ct = 0; ct < 4; ct++)
        bk[ct] = loadB(&Klds[(ct * 16 + n16) * 136 + ks * 32], hi);
#pragma unroll
      for (int ct = 0; ct < 4; ct++)
        sc[ct] = wmma_bf16(aq[ks], bk[ct], sc[ct]);
    }
    // mask + online softmax (rows r+8*hi; cols striped over 16 lanes x 4 tiles)
    float rm[8];
#pragma unroll
    for (int r = 0; r < 8; r++) rm[r] = -3.0e38f;
#pragma unroll
    for (int ct = 0; ct < 4; ct++)
#pragma unroll
      for (int r = 0; r < 8; r++) {
        float v = sc[ct][r] +
                  mask[(size_t)(qr0 + r + 8 * hi) * S + kv0 + ct * 16 + n16];
        sc[ct][r] = v;
        rm[r] = fmaxf(rm[r], v);
      }
#pragma unroll
    for (int r = 0; r < 8; r++) {
#pragma unroll
      for (int mk = 8; mk >= 1; mk >>= 1)
        rm[r] = fmaxf(rm[r], __shfl_xor(rm[r], mk, 32));
      float mn = fmaxf(mrow[r], rm[r]);
      float scale = __expf(mrow[r] - mn);
      mrow[r] = mn;
      lrow[r] *= scale;
#pragma unroll
      for (int dt = 0; dt < 8; dt++) o[dt][r] *= scale;
      float rs = 0.f;
#pragma unroll
      for (int ct = 0; ct < 4; ct++) {
        float p = __expf(sc[ct][r] - mn);
        sc[ct][r] = p;
        rs += p;
      }
#pragma unroll
      for (int mk = 8; mk >= 1; mk >>= 1) rs += __shfl_xor(rs, mk, 32);
      lrow[r] += rs;
    }
    // P (C-layout) -> wave-private LDS -> A-fragments; DS ops in-order per wave
#pragma unroll
    for (int ct = 0; ct < 4; ct++)
#pragma unroll
      for (int r = 0; r < 8; r++)
        Pw[(r + 8 * hi) * 72 + ct * 16 + n16] = f2bf(sc[ct][r]);
#pragma unroll
    for (int ks2 = 0; ks2 < 2; ks2++) {
      FragBF ap;
      const unsigned short* pp = Pw + n16 * 72 + ks2 * 32;
      ap.q[0] = *(const u32x4*)(pp + hi * 8);
      ap.q[1] = *(const u32x4*)(pp + 16 + hi * 8);
      FragBF bv[8];
#pragma unroll
      for (int dt = 0; dt < 8; dt++)
        bv[dt] = loadB(&Vtlds[(dt * 16 + n16) * 72 + ks2 * 32], hi);
#pragma unroll
      for (int dt = 0; dt < 8; dt++)
        o[dt] = wmma_bf16(ap, bv[dt], o[dt]);
    }
    __syncthreads();
  }
#pragma unroll
  for (int dt = 0; dt < 8; dt++)
#pragma unroll
    for (int r = 0; r < 8; r++) {
      int srow = qr0 + r + 8 * hi;
      float val = o[dt][r] / lrow[r];
      O[((size_t)b * S + srow) * H + h * HD + dt * 16 + n16] = f2bf(val);
    }
}

extern "C" void kernel_launch(void* const* d_in, const int* in_sizes, int n_in,
                              void* d_out, int out_size, void* d_ws, size_t ws_size,
                              hipStream_t stream) {
  (void)in_sizes; (void)n_in; (void)out_size; (void)ws_size;
  const float* hidden = (const float*)d_in[0];
  const float* mask   = (const float*)d_in[1];
  const float* wq     = (const float*)d_in[2];
  const float* wk     = (const float*)d_in[3];
  const float* wv     = (const float*)d_in[4];
  const float* wo     = (const float*)d_in[5];

  const int S = 2048, H = 2048, B = 2, M = B * S;
  unsigned short* ws    = (unsigned short*)d_ws;
  unsigned short* x_bf  = ws;
  unsigned short* wq_bf = x_bf + (size_t)M * H;
  unsigned short* wk_bf = wq_bf + (size_t)H * H;
  unsigned short* wv_bf = wk_bf + (size_t)H * H;
  unsigned short* wo_bf = wv_bf + (size_t)H * H;
  unsigned short* q_bf  = wo_bf + (size_t)H * H;
  unsigned short* k_bf  = q_bf + (size_t)M * H;
  unsigned short* vT_bf = k_bf + (size_t)M * H;
  unsigned short* ao_bf = vT_bf + (size_t)M * H;

  const int nX8 = (M * H) / 8, nW8 = (H * H) / 8;
  svarog_cvt_bf16<<<(nX8 + 255) / 256, 256, 0, stream>>>(hidden, x_bf, nX8);
  svarog_cvt_bf16<<<(nW8 + 255) / 256, 256, 0, stream>>>(wq, wq_bf, nW8);
  svarog_cvt_bf16<<<(nW8 + 255) / 256, 256, 0, stream>>>(wk, wk_bf, nW8);
  svarog_cvt_bf16<<<(nW8 + 255) / 256, 256, 0, stream>>>(wv, wv_bf, nW8);
  svarog_cvt_bf16<<<(nW8 + 255) / 256, 256, 0, stream>>>(wo, wo_bf, nW8);

  dim3 g(H / 128, M / 256);  // (16, 16)
  svarog_gemm_wmma<<<g, 256, 0, stream>>>(x_bf, wq_bf, q_bf, nullptr, 1);
  svarog_gemm_wmma<<<g, 256, 0, stream>>>(x_bf, wk_bf, k_bf, nullptr, 2);
  svarog_gemm_wmma<<<g, 256, 0, stream>>>(x_bf, wv_bf, vT_bf, nullptr, 3);

  svarog_flash_wmma<<<dim3(S / 128, B * 16), 256, 0, stream>>>(
      q_bf, k_bf, vT_bf, mask, ao_bf);

  svarog_gemm_wmma<<<g, 256, 0, stream>>>(ao_bf, wo_bf, nullptr, (float*)d_out, 0);
}
